// gat_48344151884424
// MI455X (gfx1250) — matrix operand order repaired and checked by reading, not compiled, not run on hardware
//
#include <hip/hip_runtime.h>
#include <hip/hip_bf16.h>
#include <cstdint>

#define N_HEADS  2
#define HIDF     64
#define INF_     128
#define OUTF     32
#define NHID     (N_HEADS * HIDF)   // 128
#define NEG_SLOPE 0.2f

typedef __attribute__((ext_vector_type(16))) __bf16 v16bf;
typedef __attribute__((ext_vector_type(8)))  float  v8f;

// ---------- helpers: monotone float<->uint encoding for atomic float max ----------
__device__ __forceinline__ unsigned enc_f32(float f) {
    unsigned u = __float_as_uint(f);
    return (u & 0x80000000u) ? ~u : (u | 0x80000000u);
}
__device__ __forceinline__ float dec_f32(unsigned u) {
    unsigned b = (u & 0x80000000u) ? (u ^ 0x80000000u) : ~u;
    return __uint_as_float(b);
}

// ---------- bf16 conversion of x ----------
__global__ void cvt_x(const float* __restrict__ x, __bf16* __restrict__ xbf, int n) {
    int i = blockIdx.x * blockDim.x + threadIdx.x;
    if (i < n) xbf[i] = (__bf16)x[i];
}

// ---------- W [K=128][N=128] -> transposed bf16 wt[n*128 + k] ----------
__global__ void cvt_wt(const float* __restrict__ W, __bf16* __restrict__ wt) {
    int i = blockIdx.x * blockDim.x + threadIdx.x;  // 16384
    int k = i >> 7, n = i & 127;
    wt[n * INF_ + k] = (__bf16)W[i];
}

// ---------- zero / -inf init of accumulation state ----------
__global__ void init_buffers(float* __restrict__ rst, float* __restrict__ ssum,
                             unsigned* __restrict__ mmax, int nNodes) {
    int i = blockIdx.x * blockDim.x + threadIdx.x;
    if (i < nNodes * NHID) rst[i] = 0.0f;
    if (i < nNodes * N_HEADS) { ssum[i] = 0.0f; mmax[i] = 0u; }  // 0u decodes to NaN -> "empty"
}

// ---------- projection GEMM via v_wmma_f32_16x16x32_bf16 ----------
// grid: nNodes/16 blocks of 256 threads (8 waves). wave w -> N-tile n0 = 16*w.
__global__ void __launch_bounds__(256) proj_wmma(const __bf16* __restrict__ xbf,
                                                 const __bf16* __restrict__ wt,
                                                 float* __restrict__ feat) {
    const int lane = threadIdx.x & 31;
    const int wave = threadIdx.x >> 5;
    const int m0   = blockIdx.x * 16;
    const int n0   = wave * 16;
    const int hi   = lane >> 4;   // 0: lanes 0-15, 1: lanes 16-31
    const int lo   = lane & 15;

    union { v16bf v; uint32_t u[8]; } a, b;
    v8f c = {};

    const int arow = m0 + lo;     // A row for this lane
    const int bcol = n0 + lo;     // B column for this lane

    for (int kb = 0; kb < INF_; kb += 32) {
        // A 16x32 bf16 fragment: lanes<16 hold K {0..7,16..23}, lanes>=16 hold K {8..15,24..31}
        const uint32_t* ap = (const uint32_t*)(xbf + (size_t)arow * INF_ + kb + (hi << 3));
        // B 32x16 bf16 fragment (wt is column-major of W): lanes<16 hold K {0..7,16..23}, lanes>=16 hold K {8..15,24..31}
        const uint32_t* bp = (const uint32_t*)(wt + (size_t)bcol * INF_ + kb + (hi << 3));
#pragma unroll
        for (int v = 0; v < 4; ++v) {
            a.u[v]     = ap[v];       // K pairs 0,2,4,6 (relative)
            a.u[v + 4] = ap[v + 8];   // K pairs 16,18,20,22 (relative)
            b.u[v]     = bp[v];       // K pairs 0..7 contiguous
            b.u[v + 4] = bp[v + 8];
        }
        c = __builtin_amdgcn_wmma_f32_16x16x32_bf16(false, a.v, false, b.v,
                                                    (short)0, c, false, false);
    }
    // C layout: VGPR r, lanes<16 -> M=r, lanes>=16 -> M=8+r; N = n0+lo
#pragma unroll
    for (int r = 0; r < 8; ++r)
        feat[(size_t)(m0 + r + (hi << 3)) * NHID + n0 + lo] = c[r];
}

// ---------- el/er per (node, head) ----------
__global__ void attn_dots(const float* __restrict__ feat, const float* __restrict__ al,
                          const float* __restrict__ ar, float* __restrict__ el,
                          float* __restrict__ er, int nNodes) {
    int i = blockIdx.x * blockDim.x + threadIdx.x;
    if (i >= nNodes * N_HEADS) return;
    int h = i & 1, nd = i >> 1;
    const float* f  = feat + (size_t)nd * NHID + h * HIDF;
    const float* wl = al + h * HIDF;
    const float* wr = ar + h * HIDF;
    float sl = 0.f, sr = 0.f;
#pragma unroll 4
    for (int j = 0; j < HIDF; ++j) { float v = f[j]; sl += v * wl[j]; sr += v * wr[j]; }
    el[i] = sl; er[i] = sr;
}

// ---------- pass 1: segment max of leaky-relu logits ----------
__global__ void edge_max(const int* __restrict__ src, const int* __restrict__ dst,
                         const float* __restrict__ el, const float* __restrict__ er,
                         unsigned* __restrict__ mmax, int nEdges) {
    int e = blockIdx.x * blockDim.x + threadIdx.x;
    if (e >= nEdges) return;
    int s = src[e], d = dst[e];
#pragma unroll
    for (int h = 0; h < N_HEADS; ++h) {
        float v = el[s * 2 + h] + er[d * 2 + h];
        v = v > 0.f ? v : NEG_SLOPE * v;
        atomicMax(&mmax[d * 2 + h], enc_f32(v));
    }
}

// ---------- pass 2: exp(e - m), segment sum ----------
__global__ void edge_exp(const int* __restrict__ src, const int* __restrict__ dst,
                         const float* __restrict__ el, const float* __restrict__ er,
                         const unsigned* __restrict__ mmax, float* __restrict__ exbuf,
                         float* __restrict__ ssum, int nEdges) {
    int e = blockIdx.x * blockDim.x + threadIdx.x;
    if (e >= nEdges) return;
    int s = src[e], d = dst[e];
#pragma unroll
    for (int h = 0; h < N_HEADS; ++h) {
        float v = el[s * 2 + h] + er[d * 2 + h];
        v = v > 0.f ? v : NEG_SLOPE * v;
        float m = dec_f32(mmax[d * 2 + h]);
        if (!isfinite(m)) m = 0.f;
        float ex = __expf(v - m);
        exbuf[(size_t)e * 2 + h] = ex;
        atomicAdd(&ssum[d * 2 + h], ex);
    }
}

// ---------- pass 3: weighted scatter-add aggregation; one wave32 per edge ----------
__global__ void __launch_bounds__(256) aggregate(const int* __restrict__ src,
                                                 const int* __restrict__ dst,
                                                 const float* __restrict__ exbuf,
                                                 const float* __restrict__ ssum,
                                                 const float* __restrict__ feat,
                                                 float* __restrict__ rst, int nEdges) {
    int lane = threadIdx.x & 31;
    int e = blockIdx.x * 8 + (threadIdx.x >> 5);
    if (e >= nEdges) return;
    int s = src[e], d = dst[e];
    int h = lane >> 4;  // features [lane*4, lane*4+3] all belong to head lane>>4
    float alpha = exbuf[(size_t)e * 2 + h] / fmaxf(ssum[d * 2 + h], 1e-9f);
    const float4 fv = *(const float4*)(feat + (size_t)s * NHID + lane * 4);
    float* o = rst + (size_t)d * NHID + lane * 4;
    atomicAdd(o + 0, alpha * fv.x);
    atomicAdd(o + 1, alpha * fv.y);
    atomicAdd(o + 2, alpha * fv.z);
    atomicAdd(o + 3, alpha * fv.w);
}

// ---------- bias + relu + head-mean + final linear; one wave32 per node ----------
__global__ void __launch_bounds__(256) final_mlp(const float* __restrict__ rst,
                                                 const float* __restrict__ bias,
                                                 const float* __restrict__ fcw,
                                                 const float* __restrict__ fcb,
                                                 float* __restrict__ out, int nNodes) {
    __shared__ float hmean[8][HIDF];
    int lane = threadIdx.x & 31;
    int wv   = threadIdx.x >> 5;
    int n    = blockIdx.x * 8 + wv;
    if (n < nNodes) {
#pragma unroll
        for (int j = 0; j < 2; ++j) {
            int f = lane + j * 32;
            float a = rst[(size_t)n * NHID + f]      + bias[f];
            float b = rst[(size_t)n * NHID + 64 + f] + bias[64 + f];
            hmean[wv][f] = 0.5f * (fmaxf(a, 0.f) + fmaxf(b, 0.f));
        }
    }
    __syncthreads();
    if (n < nNodes) {
        float acc = fcb[lane];
#pragma unroll 4
        for (int f = 0; f < HIDF; ++f) acc += hmean[wv][f] * fcw[f * OUTF + lane];
        out[(size_t)n * OUTF + lane] = acc;
    }
}

extern "C" void kernel_launch(void* const* d_in, const int* in_sizes, int n_in,
                              void* d_out, int out_size, void* d_ws, size_t ws_size,
                              hipStream_t stream) {
    const float* x    = (const float*)d_in[0];
    const int*   src  = (const int*)d_in[1];
    const int*   dst  = (const int*)d_in[2];
    const float* W    = (const float*)d_in[3];
    const float* al   = (const float*)d_in[4];
    const float* ar   = (const float*)d_in[5];
    const float* bias = (const float*)d_in[6];
    const float* fcw  = (const float*)d_in[7];
    const float* fcb  = (const float*)d_in[8];
    float* out = (float*)d_out;

    const int nNodes = in_sizes[0] / INF_;   // 50000
    const int nEdges = in_sizes[1];          // 1600000

    char* ws = (char*)d_ws;
    size_t off = 0;
    auto alloc = [&](size_t bytes) -> char* {
        char* p = ws + off;
        off = (off + bytes + 255) & ~(size_t)255;
        return p;
    };
    float*    feat  = (float*)alloc((size_t)nNodes * NHID * sizeof(float));
    float*    rst   = (float*)alloc((size_t)nNodes * NHID * sizeof(float));
    __bf16*   xbf   = (__bf16*)alloc((size_t)nNodes * INF_ * sizeof(__bf16));
    __bf16*   wtbf  = (__bf16*)alloc((size_t)INF_ * NHID * sizeof(__bf16));
    float*    el    = (float*)alloc((size_t)nNodes * N_HEADS * sizeof(float));
    float*    er    = (float*)alloc((size_t)nNodes * N_HEADS * sizeof(float));
    unsigned* mmax  = (unsigned*)alloc((size_t)nNodes * N_HEADS * sizeof(unsigned));
    float*    ssum  = (float*)alloc((size_t)nNodes * N_HEADS * sizeof(float));
    float*    exbuf = (float*)alloc((size_t)nEdges * N_HEADS * sizeof(float));
    (void)ws_size; (void)n_in; (void)out_size;

    const int txc = nNodes * INF_;
    cvt_x<<<(txc + 255) / 256, 256, 0, stream>>>(x, xbf, txc);
    cvt_wt<<<(INF_ * NHID + 255) / 256, 256, 0, stream>>>(W, wtbf);
    init_buffers<<<((size_t)nNodes * NHID + 255) / 256, 256, 0, stream>>>(rst, ssum, mmax, nNodes);
    proj_wmma<<<nNodes / 16, 256, 0, stream>>>(xbf, wtbf, feat);
    attn_dots<<<(nNodes * N_HEADS + 255) / 256, 256, 0, stream>>>(feat, al, ar, el, er, nNodes);
    edge_max<<<(nEdges + 255) / 256, 256, 0, stream>>>(src, dst, el, er, mmax, nEdges);
    edge_exp<<<(nEdges + 255) / 256, 256, 0, stream>>>(src, dst, el, er, mmax, exbuf, ssum, nEdges);
    aggregate<<<(nEdges + 7) / 8, 256, 0, stream>>>(src, dst, exbuf, ssum, feat, rst, nEdges);
    final_mlp<<<(nNodes + 7) / 8, 256, 0, stream>>>(rst, bias, fcw, fcb, out, nNodes);
}
